// PatchcoreModel_75479755260539
// MI455X (gfx1250) — compile-verified, matrix operand order checked by reading, not run on hardware
//
#include <hip/hip_runtime.h>
#include <math.h>

// ---------------- problem constants ----------------
#define BATCH   8
#define C2      512
#define C3      1024
#define HW      28          // layer2 spatial
#define HW3     14          // layer3 spatial
#define NROWS   (BATCH * HW * HW)   // 6272 embedding rows
#define DDIM    1536
#define MROWS   16384
#define OUTHW   224
#define AMAPN   (BATCH * OUTHW * OUTHW)   // 401408

typedef unsigned short u16;
typedef __attribute__((ext_vector_type(16))) __bf16 v16bf;
typedef __attribute__((ext_vector_type(8)))  float  v8f;
typedef __attribute__((__vector_size__(16))) int vi4;   // 16B payload unit for async LDS copies

#define AS1 __attribute__((address_space(1)))
#define AS3 __attribute__((address_space(3)))

// gfx1250 async global->LDS copy (ASYNCcnt-tracked), with portable fallback.
#if defined(__has_builtin)
#if __has_builtin(__builtin_amdgcn_global_load_async_to_lds_b128)
#define HAVE_ASYNC_LDS 1
#endif
#endif

#ifndef HAVE_ASYNC_LDS
#define HAVE_ASYNC_LDS 0
#endif

#if HAVE_ASYNC_LDS
#if __has_builtin(__builtin_amdgcn_s_wait_asynccnt)
#define ASYNC_WAIT() __builtin_amdgcn_s_wait_asynccnt(0)
#else
#define ASYNC_WAIT() asm volatile("s_wait_asynccnt 0x0" ::: "memory")
#endif
// builtin signature (from compiler diagnostic): (AS1 vi4* src, AS3 vi4* dst, imm off, imm cpol)
#define ASYNC_CP16(gptr, lptr, idx) \
    __builtin_amdgcn_global_load_async_to_lds_b128((gptr) + (idx), (lptr) + (idx), 0, 0)
#else
#define ASYNC_WAIT() ((void)0)
#endif

union FragBF {
    v16bf v;
    uint4 q[2];
};

__device__ __forceinline__ u16 f2bf(float f) {
    unsigned int u = __float_as_uint(f);
    unsigned int r = u + 0x7FFFu + ((u >> 16) & 1u);   // round-to-nearest-even
    return (u16)(r >> 16);
}

// ---------------- K1: avgpool3x3 + nearest-exact upsample + concat -> embedding ----
__global__ void k_embed(const float* __restrict__ f2, const float* __restrict__ f3,
                        float* __restrict__ embF, u16* __restrict__ embB) {
    int idx = blockIdx.x * 256 + threadIdx.x;
    if (idx >= NROWS * DDIM) return;
    int n = idx / DDIM;
    int d = idx - n * DDIM;
    int b = n / (HW * HW);
    int p = n - b * (HW * HW);
    int y = p / HW;
    int x = p - y * HW;
    float s = 0.0f;
    if (d < C2) {
        const float* base = f2 + (size_t)(b * C2 + d) * (HW * HW);
        #pragma unroll
        for (int dy = -1; dy <= 1; ++dy) {
            int yy = y + dy;
            if (yy < 0 || yy >= HW) continue;
            #pragma unroll
            for (int dx = -1; dx <= 1; ++dx) {
                int xx = x + dx;
                if (xx < 0 || xx >= HW) continue;
                s += base[yy * HW + xx];
            }
        }
    } else {
        int c = d - C2;
        int ys = y >> 1, xs = x >> 1;   // nearest-exact for 2x: floor((i+0.5)/2)
        const float* base = f3 + (size_t)(b * C3 + c) * (HW3 * HW3);
        #pragma unroll
        for (int dy = -1; dy <= 1; ++dy) {
            int yy = ys + dy;
            if (yy < 0 || yy >= HW3) continue;
            #pragma unroll
            for (int dx = -1; dx <= 1; ++dx) {
                int xx = xs + dx;
                if (xx < 0 || xx >= HW3) continue;
                s += base[yy * HW3 + xx];
            }
        }
    }
    float v = s * (1.0f / 9.0f);
    embF[idx] = v;
    embB[idx] = f2bf(v);
}

// ---------------- K1b: row squared norms (exact f32) ----------------
__global__ void k_rownorm(const float* __restrict__ x, float* __restrict__ out) {
    int row = blockIdx.x, tid = threadIdx.x;
    __shared__ float red[256];
    const float* p = x + (size_t)row * DDIM;
    float s = 0.0f;
    for (int k = tid; k < DDIM; k += 256) s += p[k] * p[k];
    red[tid] = s;
    __syncthreads();
    for (int st = 128; st > 0; st >>= 1) {
        if (tid < st) red[tid] += red[tid + st];
        __syncthreads();
    }
    if (tid == 0) out[row] = red[0];
}

// ---------------- K2: bank -> bf16 + exact norms ----------------
__global__ void k_bankprep(const float* __restrict__ bank, u16* __restrict__ bankB,
                           float* __restrict__ bm2) {
    int row = blockIdx.x, tid = threadIdx.x;
    __shared__ float red[256];
    const float* p = bank + (size_t)row * DDIM;
    u16* q = bankB + (size_t)row * DDIM;
    float s = 0.0f;
    for (int k = tid; k < DDIM; k += 256) {
        float v = p[k];
        q[k] = f2bf(v);
        s += v * v;
    }
    red[tid] = s;
    __syncthreads();
    for (int st = 128; st > 0; st >>= 1) {
        if (tid < st) red[tid] += red[tid + st];
        __syncthreads();
    }
    if (tid == 0) bm2[row] = red[0];
}

// ---------------- K3: fused WMMA 1-NN search ----------------
// WG = 256 threads (8 wave32). One 16-row embedding tile per WG, resident in LDS
// (48KB bf16). M streamed in 128-row blocks; B chunks (128x128 bf16) double-
// buffered in LDS via async global->LDS copies issued one chunk ahead so L2
// latency hides under the WMMAs. Two f32 accumulators break the WMMA RAW chain.
// Epilogue fuses d2 = |a|^2 + |b|^2 - 2ab with per-lane running min/argmin.

#define BS_ELEMS (128 * 128)
#define CH_PER_MB (DDIM / 128)                      // 12
#define NCHUNK ((MROWS / 128) * CH_PER_MB)          // 1536

__device__ __forceinline__ void stageB(const u16* __restrict__ bankB, u16* smemBase,
                                       int c, int bufSel, int tid) {
    int mb = (c / CH_PER_MB) * 128;
    int kcE = (c - (c / CH_PER_MB) * CH_PER_MB) * 128;
    int r = tid >> 1;
    int co = (tid & 1) * 64;                        // 64 bf16 = 128B per thread
    const u16* src = bankB + (size_t)(mb + r) * DDIM + kcE + co;
    u16* dstG = smemBase + 16 * DDIM + bufSel * BS_ELEMS + r * 128 + co;
#if HAVE_ASYNC_LDS
    AS1 vi4* g = (AS1 vi4*)(void*)(u16*)src;
    AS3 vi4* l = (AS3 vi4*)(void*)dstG;
    #pragma unroll
    for (int j = 0; j < 8; ++j) ASYNC_CP16(g, l, j);
#else
    uint4 t[8];
    const uint4* s4 = (const uint4*)src;
    uint4* d4 = (uint4*)dstG;
    #pragma unroll
    for (int j = 0; j < 8; ++j) t[j] = s4[j];      // all loads first -> MLP
    #pragma unroll
    for (int j = 0; j < 8; ++j) d4[j] = t[j];
    __builtin_prefetch((const char*)s4 + 256, 0, 0);
#endif
}

#define LOADF(fr, base, ks)                                   \
    do {                                                      \
        (fr).q[0] = *(const uint4*)((base) + (ks));           \
        (fr).q[1] = *(const uint4*)((base) + (ks) + 16);      \
    } while (0)

#define WMMA_BF16(acc, a, b)                                  \
    __builtin_amdgcn_wmma_f32_16x16x32_bf16(false, (a).v, false, (b).v, (short)0, (acc), false, false)

__global__ void k_nnsearch(const u16* __restrict__ embB, const u16* __restrict__ bankB,
                           const float* __restrict__ an2, const float* __restrict__ bm2,
                           float* __restrict__ ps, int* __restrict__ loc) {
    extern __shared__ __align__(16) u16 smem[];
    u16* As = smem;                                 // 16 x 1536 bf16 = 48KB
    u16* Bs = smem + 16 * DDIM;                     // 2 x (128 x 128) bf16 = 64KB
    float* redV = (float*)(smem + 16 * DDIM + 2 * BS_ELEMS);   // 8 waves x 16 rows
    int*   redI = (int*)(redV + 128);

    const int tid = threadIdx.x;
    const int lane = tid & 31;
    const int wv = tid >> 5;
    const int n0 = blockIdx.x * 16;
    const int col = lane & 15;
    const int half8 = (lane >> 4) << 3;   // K sub-offset for A/B frags AND +8 row offset for C

    // stage A tile (16 contiguous rows -> flat 48KB copy)
    {
        const u16* srcA = embB + (size_t)n0 * DDIM;
#if HAVE_ASYNC_LDS
        AS1 vi4* g = (AS1 vi4*)(void*)(u16*)srcA;
        AS3 vi4* l = (AS3 vi4*)(void*)As;
        for (int i = tid; i < (16 * DDIM) / 8; i += 256) ASYNC_CP16(g, l, i);
#else
        const uint4* src = (const uint4*)srcA;
        uint4* dst = (uint4*)As;
        for (int i = tid; i < (16 * DDIM) / 8; i += 256) dst[i] = src[i];
#endif
    }
    // stage first B chunk into buffer 0
    stageB(bankB, smem, 0, 0, tid);
    ASYNC_WAIT();
    __syncthreads();

    float an2r[8];
    #pragma unroll
    for (int v = 0; v < 8; ++v) an2r[v] = an2[n0 + v + half8];

    float bestD[8];
    int bestI[8];
    #pragma unroll
    for (int v = 0; v < 8; ++v) { bestD[v] = 3.4e38f; bestI[v] = 0; }

    v8f acc0 = {}, acc1 = {};
    for (int c = 0; c < NCHUNK; ++c) {
        const int buf = c & 1;
        // kick off next chunk's async copy into the other buffer (overlaps compute)
        if (c + 1 < NCHUNK) stageB(bankB, smem, c + 1, buf ^ 1, tid);

        const int mbBase = (c / CH_PER_MB) * 128;
        const int kcE = (c - (c / CH_PER_MB) * CH_PER_MB) * 128;
        const u16* ap = As + col * DDIM + kcE + half8;
        const u16* bp = Bs + buf * BS_ELEMS + ((wv << 4) + col) * 128 + half8;

        FragBF a0, b0, a1, b1;
        LOADF(a0, ap, 0);
        LOADF(b0, bp, 0);
        LOADF(a1, ap, 32);
        LOADF(b1, bp, 32);
        acc0 = WMMA_BF16(acc0, a0, b0);
        LOADF(a0, ap, 64);
        LOADF(b0, bp, 64);
        acc1 = WMMA_BF16(acc1, a1, b1);
        LOADF(a1, ap, 96);
        LOADF(b1, bp, 96);
        acc0 = WMMA_BF16(acc0, a0, b0);
        acc1 = WMMA_BF16(acc1, a1, b1);

        if (kcE == DDIM - 128) {   // finished full-D accumulation for this m block
            const int m = mbBase + (wv << 4) + col;
            const float bm = bm2[m];
            #pragma unroll
            for (int v = 0; v < 8; ++v) {
                float d2 = an2r[v] + bm - 2.0f * (acc0[v] + acc1[v]);
                if (d2 < bestD[v]) { bestD[v] = d2; bestI[v] = m; }
            }
            v8f zz = {};
            acc0 = zz;
            acc1 = zz;
        }
        ASYNC_WAIT();      // our next-chunk copies have landed in LDS
        __syncthreads();   // everyone done reading cur + everyone's copies visible
    }

    // reduce across the 16 m-columns within each 16-lane half
    #pragma unroll
    for (int v = 0; v < 8; ++v) {
        float d = bestD[v];
        int idx = bestI[v];
        #pragma unroll
        for (int off = 8; off > 0; off >>= 1) {
            float od = __shfl_xor(d, off, 32);
            int   oi = __shfl_xor(idx, off, 32);
            if (od < d || (od == d && oi < idx)) { d = od; idx = oi; }
        }
        if (col == 0) {
            int n = v + half8;
            redV[wv * 16 + n] = d;
            redI[wv * 16 + n] = idx;
        }
    }
    __syncthreads();
    if (tid < 16) {
        float d = redV[tid];
        int idx = redI[tid];
        for (int ww = 1; ww < 8; ++ww) {
            float od = redV[ww * 16 + tid];
            int oi = redI[ww * 16 + tid];
            if (od < d || (od == d && oi < idx)) { d = od; idx = oi; }
        }
        ps[n0 + tid] = sqrtf(fmaxf(d, 0.0f));
        loc[n0 + tid] = idx;
    }
}

// ---------------- K4a: per-batch argmax of patch scores ----------------
__global__ void k_batch_max(const float* __restrict__ ps, const int* __restrict__ loc,
                            float* __restrict__ score, int* __restrict__ maxIdx,
                            int* __restrict__ nnIdx) {
    int b = blockIdx.x, tid = threadIdx.x;
    __shared__ float red[256];
    __shared__ int redi[256];
    float bv = -3.4e38f;
    int bi = 1 << 30;
    for (int i = tid; i < HW * HW; i += 256) {
        float v = ps[b * (HW * HW) + i];
        if (v > bv || (v == bv && i < bi)) { bv = v; bi = i; }
    }
    red[tid] = bv; redi[tid] = bi;
    __syncthreads();
    for (int s = 128; s > 0; s >>= 1) {
        if (tid < s) {
            float ov = red[tid + s]; int oi = redi[tid + s];
            if (ov > red[tid] || (ov == red[tid] && oi < redi[tid])) { red[tid] = ov; redi[tid] = oi; }
        }
        __syncthreads();
    }
    if (tid == 0) {
        score[b] = red[0];
        maxIdx[b] = redi[0];
        nnIdx[b] = loc[b * (HW * HW) + redi[0]];
    }
}

// ---------------- K4b: d2(nn_sample[b], bank[m]) for all m ----------------
__global__ void k_nn_dist(const float* __restrict__ bank, const float* __restrict__ bm2,
                          const int* __restrict__ nnIdx, float* __restrict__ d2nn) {
    int b = blockIdx.y;
    int g = threadIdx.x >> 5;
    int lane = threadIdx.x & 31;
    int m = blockIdx.x * 8 + g;
    int nn = nnIdx[b];
    const float* pa = bank + (size_t)nn * DDIM;
    const float* pb = bank + (size_t)m * DDIM;
    float dot = 0.0f;
    for (int k = lane; k < DDIM; k += 32) dot += pa[k] * pb[k];
    #pragma unroll
    for (int off = 16; off > 0; off >>= 1) dot += __shfl_xor(dot, off, 32);
    if (lane == 0) d2nn[(size_t)b * MROWS + m] = bm2[nn] + bm2[m] - 2.0f * dot;
}

// ---------------- K4c: top-9 selection + softmax re-weighting ----------------
__global__ void k_finalize(const float* __restrict__ d2nn, const float* __restrict__ embF,
                           const float* __restrict__ bank, const float* __restrict__ score,
                           const int* __restrict__ maxIdx, float* __restrict__ outScore) {
    int b = blockIdx.x, tid = threadIdx.x;
    const float* d2 = d2nn + (size_t)b * MROWS;
    __shared__ float red[256];
    __shared__ int redi[256];
    __shared__ int sel[9];
    __shared__ float sdist[9];
    for (int kk = 0; kk < 9; ++kk) {
        float bv = 3.4e38f;
        int bi = 1 << 30;
        for (int m = tid; m < MROWS; m += 256) {
            bool skip = false;
            for (int j = 0; j < kk; ++j) if (sel[j] == m) skip = true;
            if (skip) continue;
            float v = d2[m];
            if (v < bv || (v == bv && m < bi)) { bv = v; bi = m; }
        }
        red[tid] = bv; redi[tid] = bi;
        __syncthreads();
        for (int s = 128; s > 0; s >>= 1) {
            if (tid < s) {
                float ov = red[tid + s]; int oi = redi[tid + s];
                if (ov < red[tid] || (ov == red[tid] && oi < redi[tid])) { red[tid] = ov; redi[tid] = oi; }
            }
            __syncthreads();
        }
        if (tid == 0) sel[kk] = redi[0];
        __syncthreads();
    }
    const float* mp = embF + ((size_t)b * (HW * HW) + maxIdx[b]) * DDIM;
    for (int j = 0; j < 9; ++j) {
        const float* sp = bank + (size_t)sel[j] * DDIM;
        float s = 0.0f;
        for (int k = tid; k < DDIM; k += 256) { float dd = mp[k] - sp[k]; s += dd * dd; }
        red[tid] = s;
        __syncthreads();
        for (int st = 128; st > 0; st >>= 1) {
            if (tid < st) red[tid] += red[tid + st];
            __syncthreads();
        }
        if (tid == 0) sdist[j] = sqrtf(fmaxf(red[0], 0.0f));
        __syncthreads();
    }
    if (tid == 0) {
        float mx = sdist[0];
        for (int j = 1; j < 9; ++j) mx = fmaxf(mx, sdist[j]);
        float se = 0.0f, e0 = 0.0f;
        for (int j = 0; j < 9; ++j) {
            float e = __expf(sdist[j] - mx);
            se += e;
            if (j == 0) e0 = e;
        }
        outScore[b] = (1.0f - e0 / se) * score[b];
    }
}

// ---------------- K5a: bilinear 28 -> 224 (half-pixel) ----------------
__global__ void k_resize(const float* __restrict__ ps, float* __restrict__ out) {
    int idx = blockIdx.x * 256 + threadIdx.x;
    if (idx >= AMAPN) return;
    int b = idx / (OUTHW * OUTHW);
    int r = idx - b * (OUTHW * OUTHW);
    int oy = r / OUTHW;
    int ox = r - oy * OUTHW;
    float sy = (oy + 0.5f) * 0.125f - 0.5f;
    float sx = (ox + 0.5f) * 0.125f - 0.5f;
    int y0 = (int)floorf(sy), x0 = (int)floorf(sx);
    float fy = sy - y0, fx = sx - x0;
    int y0c = min(max(y0, 0), HW - 1), y1c = min(max(y0 + 1, 0), HW - 1);
    int x0c = min(max(x0, 0), HW - 1), x1c = min(max(x0 + 1, 0), HW - 1);
    const float* pb = ps + b * (HW * HW);
    float v00 = pb[y0c * HW + x0c], v01 = pb[y0c * HW + x1c];
    float v10 = pb[y1c * HW + x0c], v11 = pb[y1c * HW + x1c];
    float v0 = v00 + fx * (v01 - v00);
    float v1 = v10 + fx * (v11 - v10);
    out[idx] = v0 + fy * (v1 - v0);
}

// ---------------- K5b/K5c: 33-tap Gaussian (sigma=4), reflect pad ----------------
__global__ void k_blur_v(const float* __restrict__ in, float* __restrict__ out) {
    int idx = blockIdx.x * 256 + threadIdx.x;
    if (idx >= AMAPN) return;
    int b = idx / (OUTHW * OUTHW);
    int r = idx - b * (OUTHW * OUTHW);
    int oy = r / OUTHW;
    int ox = r - oy * OUTHW;
    float tot = 0.0f, acc = 0.0f;
    for (int t = 0; t < 33; ++t) {
        float u = (t - 16) * 0.25f;
        float wgt = __expf(-0.5f * u * u);
        tot += wgt;
        int ry = oy + t - 16;
        ry = ry < 0 ? -ry : ry;
        ry = ry > OUTHW - 1 ? 2 * (OUTHW - 1) - ry : ry;
        acc += wgt * in[b * (OUTHW * OUTHW) + ry * OUTHW + ox];
    }
    out[idx] = acc / tot;
}

__global__ void k_blur_h(const float* __restrict__ in, float* __restrict__ out) {
    int idx = blockIdx.x * 256 + threadIdx.x;
    if (idx >= AMAPN) return;
    int b = idx / (OUTHW * OUTHW);
    int r = idx - b * (OUTHW * OUTHW);
    int oy = r / OUTHW;
    int ox = r - oy * OUTHW;
    float tot = 0.0f, acc = 0.0f;
    const float* row = in + b * (OUTHW * OUTHW) + oy * OUTHW;
    for (int t = 0; t < 33; ++t) {
        float u = (t - 16) * 0.25f;
        float wgt = __expf(-0.5f * u * u);
        tot += wgt;
        int rx = ox + t - 16;
        rx = rx < 0 ? -rx : rx;
        rx = rx > OUTHW - 1 ? 2 * (OUTHW - 1) - rx : rx;
        acc += wgt * row[rx];
    }
    out[idx] = acc / tot;
}

// ---------------- host launcher ----------------
extern "C" void kernel_launch(void* const* d_in, const int* in_sizes, int n_in,
                              void* d_out, int out_size, void* d_ws, size_t ws_size,
                              hipStream_t stream) {
    (void)in_sizes; (void)n_in; (void)out_size; (void)ws_size;
    const float* f2   = (const float*)d_in[0];
    const float* f3   = (const float*)d_in[1];
    const float* bank = (const float*)d_in[2];
    float* out = (float*)d_out;

    char* ws = (char*)d_ws;
    size_t off = 0;
    auto carve = [&](size_t bytes) -> void* {
        void* p = ws + off;
        off = (off + bytes + 255) & ~(size_t)255;
        return p;
    };
    float* embF  = (float*)carve((size_t)NROWS * DDIM * 4);
    u16*   embB  = (u16*)  carve((size_t)NROWS * DDIM * 2);
    u16*   bankB = (u16*)  carve((size_t)MROWS * DDIM * 2);
    float* an2   = (float*)carve((size_t)NROWS * 4);
    float* bm2   = (float*)carve((size_t)MROWS * 4);
    float* ps    = (float*)carve((size_t)NROWS * 4);
    int*   loc   = (int*)  carve((size_t)NROWS * 4);
    float* score = (float*)carve(BATCH * 4);
    int*   maxIdx= (int*)  carve(BATCH * 4);
    int*   nnIdx = (int*)  carve(BATCH * 4);
    float* d2nn  = (float*)carve((size_t)BATCH * MROWS * 4);
    float* amap0 = (float*)carve((size_t)AMAPN * 4);
    float* amap1 = (float*)carve((size_t)AMAPN * 4);

    const int smemBytes = 16 * DDIM * 2 + 2 * BS_ELEMS * 2 + 128 * 4 + 128 * 4;  // 115712
    (void)hipFuncSetAttribute(reinterpret_cast<const void*>(k_nnsearch),
                              hipFuncAttributeMaxDynamicSharedMemorySize, smemBytes);

    k_embed    <<<(NROWS * DDIM + 255) / 256, 256, 0, stream>>>(f2, f3, embF, embB);
    k_rownorm  <<<NROWS, 256, 0, stream>>>(embF, an2);
    k_bankprep <<<MROWS, 256, 0, stream>>>(bank, bankB, bm2);
    k_nnsearch <<<NROWS / 16, 256, smemBytes, stream>>>(embB, bankB, an2, bm2, ps, loc);
    k_batch_max<<<BATCH, 256, 0, stream>>>(ps, loc, score, maxIdx, nnIdx);
    k_nn_dist  <<<dim3(MROWS / 8, BATCH), 256, 0, stream>>>(bank, bm2, nnIdx, d2nn);
    k_finalize <<<BATCH, 256, 0, stream>>>(d2nn, embF, bank, score, maxIdx, out + AMAPN);
    k_resize   <<<AMAPN / 256, 256, 0, stream>>>(ps, amap0);
    k_blur_v   <<<AMAPN / 256, 256, 0, stream>>>(amap0, amap1);
    k_blur_h   <<<AMAPN / 256, 256, 0, stream>>>(amap1, out);
}
